// TransformerBlock_30477087932904
// MI455X (gfx1250) — matrix operand order repaired and checked by reading, not compiled, not run on hardware
//
#include <hip/hip_runtime.h>

// Problem constants (match reference)
#define BB 8
#define TT 1024
#define CC 512
#define HH 8

typedef unsigned short u16;
typedef unsigned int   u32;
typedef long long      ll;

typedef __attribute__((ext_vector_type(16))) __bf16 v16bf;
typedef __attribute__((ext_vector_type(8)))  float  v8f;
typedef __attribute__((ext_vector_type(4)))  unsigned int v4u;
typedef __attribute__((ext_vector_type(8)))  int    v8i;
typedef __attribute__((ext_vector_type(4)))  int    v4i;

__device__ __forceinline__ u16 f2bf(float f) {
  u32 u = __float_as_uint(f);
  u32 r = u + 0x7FFFu + ((u >> 16) & 1u);   // round-to-nearest-even
  return (u16)(r >> 16);
}
__device__ __forceinline__ float bf2f(u16 h) {
  return __uint_as_float(((u32)h) << 16);
}

// ---------------- fp32 -> bf16 convert ----------------
__global__ void k_f32_to_bf16(const float* __restrict__ in, u16* __restrict__ out, int n) {
  int i = blockIdx.x * blockDim.x + threadIdx.x;
  if (i < n) out[i] = f2bf(in[i]);
}

// ---------------- LayerNorm (row of C=512 per 256-thread block), bf16 out ----------------
__global__ void k_layernorm_bf16(const float* __restrict__ x, const float* __restrict__ g,
                                 const float* __restrict__ b, u16* __restrict__ out) {
  __shared__ float sh[256];
  int row = blockIdx.x;
  int tid = threadIdx.x;
  const float* xr = x + (size_t)row * CC;
  float v0 = xr[tid], v1 = xr[tid + 256];
  sh[tid] = v0 + v1; __syncthreads();
  for (int s = 128; s > 0; s >>= 1) { if (tid < s) sh[tid] += sh[tid + s]; __syncthreads(); }
  float mean = sh[0] * (1.0f / CC); __syncthreads();
  float d0 = v0 - mean, d1 = v1 - mean;
  sh[tid] = d0 * d0 + d1 * d1; __syncthreads();
  for (int s = 128; s > 0; s >>= 1) { if (tid < s) sh[tid] += sh[tid + s]; __syncthreads(); }
  float rstd = rsqrtf(sh[0] * (1.0f / CC) + 1e-5f);
  u16* orow = out + (size_t)row * CC;
  orow[tid]       = f2bf(d0 * rstd * g[tid]       + b[tid]);
  orow[tid + 256] = f2bf(d1 * rstd * g[tid + 256] + b[tid + 256]);
}

// ---------------- causal softmax in place over bf16 scores, rows of length T ----------------
__global__ void k_softmax_causal(u16* __restrict__ S) {
  __shared__ float sh[256];
  int row = blockIdx.x;              // over B*H*T
  int t   = row & (TT - 1);
  int tid = threadIdx.x;
  u16* sr = S + (size_t)row * TT;
  float mx = -3.4e38f;
  for (int s = tid; s <= t; s += 256) mx = fmaxf(mx, bf2f(sr[s]));
  sh[tid] = mx; __syncthreads();
  for (int s2 = 128; s2 > 0; s2 >>= 1) { if (tid < s2) sh[tid] = fmaxf(sh[tid], sh[tid + s2]); __syncthreads(); }
  mx = sh[0]; __syncthreads();
  float sum = 0.f;
  for (int s = tid; s <= t; s += 256) sum += __expf(bf2f(sr[s]) - mx);
  sh[tid] = sum; __syncthreads();
  for (int s2 = 128; s2 > 0; s2 >>= 1) { if (tid < s2) sh[tid] += sh[tid + s2]; __syncthreads(); }
  float rinv = 1.0f / sh[0]; __syncthreads();
  for (int s = tid; s < TT; s += 256)
    sr[s] = (s <= t) ? f2bf(__expf(bf2f(sr[s]) - mx) * rinv) : (u16)0;
}

// ---------------- TDM: DMA one [rows x 32] bf16 tile (row stride ld) into LDS ----------------
// D# layout per CDNA5 ISA ch.8: group0 = {count/flags, lds_addr, global_addr, type=2},
// group1 = {data_size, tensor dims, tile dims, dim0 stride}. 2D tile -> groups 2/3 zero.
// This toolchain exposes the 6-arg builtin: (u32x4, i32x8, i32x4, i32x4, i32x8, cpol).
__device__ __forceinline__ void tdm_tile_load(const u16* g, int ld, const u16* ldsdst, int rows) {
  unsigned lds_addr = (unsigned)(uintptr_t)ldsdst;          // low 32 bits = LDS byte offset
  unsigned long long ga = (unsigned long long)(uintptr_t)g;
  v4u g0 = { 1u,                                            // count = 1 valid descriptor
             lds_addr,
             (unsigned)(ga & 0xFFFFFFFFu),
             (unsigned)((ga >> 32) & 0x01FFFFFFu) | 0x80000000u };  // addr[56:32] | type=2
  unsigned td0 = (unsigned)ld;      // tensor_dim0 (row length)
  unsigned td1 = (unsigned)rows;    // tensor_dim1 (rows)
  v8i g1;
  g1[0] = 0x00010000;                                       // data_size = 2 bytes
  g1[1] = (int)((td0 & 0xFFFFu) << 16);                     // [63:48]  tensor_dim0 lo
  g1[2] = (int)((td0 >> 16) | ((td1 & 0xFFFFu) << 16));     // [79:64] td0 hi | [95:80] td1 lo
  g1[3] = (int)((td1 >> 16) | (32u << 16));                 // [111:96] td1 hi | [127:112] tile_dim0=32
  g1[4] = (int)td1;                                         // [143:128] tile_dim1 = rows
  g1[5] = (int)td0;                                         // [191:160] tensor_dim0_stride lo = ld
  g1[6] = 0;
  g1[7] = 0;
  v4i gz4 = {0, 0, 0, 0};
  v8i gz8 = {0, 0, 0, 0, 0, 0, 0, 0};
  __builtin_amdgcn_tensor_load_to_lds(g0, g1, gz4, gz4, gz8, 0);
}

// ---------------- generic batched bf16 WMMA GEMM: D[z] = A[z] * W[z]^T (+ bias/resid/relu) ----
// A: [M,K] row-major bf16, W: [N,K] row-major bf16.
// Per-z offsets: off = (z/div)*s1 + (z%div)*s2 (element units).
// Block tile 64(M) x 128(N), 8 waves (2x4), wave tile 32x32 = 2x2 wmma 16x16x32 tiles.
// K pipeline: TDM double-buffered tiles in LDS (tensorcnt), fragments via ds_load_b128.
#define F_RELU   1
#define F_OBF16  2
#define F_TRANS  4
#define F_CAUSAL 8

union Frag { v16bf v; uint4 q[2]; };

__global__ __launch_bounds__(256)
void k_gemm_bf16(const u16* __restrict__ A, int lda, int divA, ll sA1, ll sA2,
                 const u16* __restrict__ W, int ldw, int divW, ll sW1, ll sW2,
                 void* __restrict__ Out, int ldo, int divO, ll sO1, ll sO2,
                 const float* __restrict__ resid, const float* __restrict__ bias,
                 int M, int N, int K, float scale, int flags)
{
  __shared__ u16 As[2][64 * 32];    // [buf][row][k]  4 KB each
  __shared__ u16 Bs[2][128 * 32];   // [buf][row][k]  8 KB each

  int z = blockIdx.z;
  const u16* Ab = A + (ll)(z / divA) * sA1 + (ll)(z % divA) * sA2;
  const u16* Wb = W + (ll)(z / divW) * sW1 + (ll)(z % divW) * sW2;
  ll oOff = (ll)(z / divO) * sO1 + (ll)(z % divO) * sO2;

  int m0 = blockIdx.y * 64;
  int n0 = blockIdx.x * 128;
  if ((flags & F_CAUSAL) && (n0 > m0 + 63)) return;   // upper-triangle block: skip (uniform)

  int wave = threadIdx.x >> 5;
  int lane = threadIdx.x & 31;
  int wm = wave >> 2, wn = wave & 3;           // 2x4 wave grid
  int lr = lane & 15, lh = lane >> 4;

  const u16* Atile = Ab + (ll)m0 * lda;        // 64 x K
  const u16* Wtile = Wb + (ll)n0 * ldw;        // 128 x K
  bool issuer = (wave == 0);

  v8f zf = {0.f, 0.f, 0.f, 0.f, 0.f, 0.f, 0.f, 0.f};
  v8f acc[2][2] = {{zf, zf}, {zf, zf}};

  int nk = K >> 5;
  if (issuer) {                                 // prologue: tile 0 -> buffer 0
    tdm_tile_load(Atile, lda, &As[0][0], 64);
    tdm_tile_load(Wtile, ldw, &Bs[0][0], 128);
  }

  // LDS fragment rows for this wave
  int rA0 = (wm * 32 + lr) * 32;                // A rows, tile mi adds 16*32
  int rB0 = (wn * 32 + lr) * 32;                // B rows, tile ni adds 16*32

  for (int i = 0; i < nk; ++i) {
    int cur = i & 1;
    __builtin_amdgcn_s_wait_tensorcnt(0);       // issuer: tile i landed; others: no-op
    __syncthreads();                            // publish LDS tile i; prior reads of buf cur^1 done
    if (issuer && (i + 1 < nk)) {               // overlap: DMA tile i+1 while computing tile i
      int k1 = (i + 1) << 5;
      tdm_tile_load(Atile + k1, lda, &As[cur ^ 1][0], 64);
      tdm_tile_load(Wtile + k1, ldw, &Bs[cur ^ 1][0], 128);
    }

    Frag a0, a1, b0, b1;
    // A frag (16-bit A 16x32): lanes0-15 K=[0..7]+[16..23], lanes16-31 K=[8..15]+[24..31]
    a0.q[0] = *(const uint4*)&As[cur][rA0 + lh * 8];
    a0.q[1] = *(const uint4*)&As[cur][rA0 + lh * 8 + 16];
    a1.q[0] = *(const uint4*)&As[cur][rA0 + 16 * 32 + lh * 8];
    a1.q[1] = *(const uint4*)&As[cur][rA0 + 16 * 32 + lh * 8 + 16];
    // B frag (16-bit B 32x16): lanes0-15 K=[0..7]+[16..23], lanes16-31 K=[8..15]+[24..31]
    b0.q[0] = *(const uint4*)&Bs[cur][rB0 + lh * 8];
    b0.q[1] = *(const uint4*)&Bs[cur][rB0 + lh * 8 + 16];
    b1.q[0] = *(const uint4*)&Bs[cur][rB0 + 16 * 32 + lh * 8];
    b1.q[1] = *(const uint4*)&Bs[cur][rB0 + 16 * 32 + lh * 8 + 16];

    acc[0][0] = __builtin_amdgcn_wmma_f32_16x16x32_bf16(false, a0.v, false, b0.v, (short)0, acc[0][0], false, false);
    acc[0][1] = __builtin_amdgcn_wmma_f32_16x16x32_bf16(false, a0.v, false, b1.v, (short)0, acc[0][1], false, false);
    acc[1][0] = __builtin_amdgcn_wmma_f32_16x16x32_bf16(false, a1.v, false, b0.v, (short)0, acc[1][0], false, false);
    acc[1][1] = __builtin_amdgcn_wmma_f32_16x16x32_bf16(false, a1.v, false, b1.v, (short)0, acc[1][1], false, false);
  }

  // Epilogue. C/D layout: lane(0-15)=col n, VGPR r = row r (lanes16-31: row r+8).
  int mw = m0 + wm * 32, nw = n0 + wn * 32;
  for (int mi = 0; mi < 2; ++mi) {
    for (int ni = 0; ni < 2; ++ni) {
      int n  = nw + ni * 16 + lr;
      int mb = mw + mi * 16 + lh * 8;
      float bv = bias ? bias[n] : 0.0f;
      for (int r = 0; r < 8; ++r) {
        float v = acc[mi][ni][r] * scale + bv;
        if (flags & F_RELU) v = fmaxf(v, 0.0f);
        if (flags & F_TRANS) {
          ll idx = oOff + (ll)n * ldo + (mb + r);        // write D^T (used for V^T)
          ((u16*)Out)[idx] = f2bf(v);
        } else {
          ll idx = oOff + (ll)(mb + r) * ldo + n;
          if (resid) v += resid[idx];
          if (flags & F_OBF16) ((u16*)Out)[idx] = f2bf(v);
          else                 ((float*)Out)[idx] = v;
        }
      }
    }
  }
}

// ------------------------------------------------------------------
extern "C" void kernel_launch(void* const* d_in, const int* in_sizes, int n_in,
                              void* d_out, int out_size, void* d_ws, size_t ws_size,
                              hipStream_t stream) {
  (void)in_sizes; (void)n_in; (void)out_size; (void)ws_size;
  const float* x   = (const float*)d_in[0];
  const float* Wq  = (const float*)d_in[1];
  const float* Wk  = (const float*)d_in[2];
  const float* Wv  = (const float*)d_in[3];
  const float* Wo  = (const float*)d_in[4];
  const float* bo  = (const float*)d_in[5];
  const float* W1  = (const float*)d_in[6];
  const float* b1  = (const float*)d_in[7];
  const float* W2  = (const float*)d_in[8];
  const float* b2  = (const float*)d_in[9];
  const float* g1  = (const float*)d_in[10];
  const float* be1 = (const float*)d_in[11];
  const float* g2  = (const float*)d_in[12];
  const float* be2 = (const float*)d_in[13];
  float* out = (float*)d_out;

  // ---- workspace carve-up (bf16 buffers) ----
  char* ws = (char*)d_ws;
  size_t off = 0;
  auto alloc = [&](size_t bytes) -> void* {
    void* p = ws + off;
    off += (bytes + 255) & ~(size_t)255;
    return p;
  };
  u16* hb   = (u16*)alloc((size_t)BB * TT * CC * 2);          // ln1(x) bf16
  u16* wqb  = (u16*)alloc((size_t)HH * CC * CC * 2);
  u16* wkb  = (u16*)alloc((size_t)HH * CC * CC * 2);
  u16* wvb  = (u16*)alloc((size_t)HH * CC * CC * 2);
  u16* wob  = (u16*)alloc((size_t)CC * HH * CC * 2);
  u16* w1b  = (u16*)alloc((size_t)4 * CC * CC * 2);
  u16* w2b  = (u16*)alloc((size_t)CC * 4 * CC * 2);
  u16* Qb   = (u16*)alloc((size_t)BB * HH * TT * CC * 2);     // [B,H,T,C]
  u16* Kb   = (u16*)alloc((size_t)BB * HH * TT * CC * 2);     // [B,H,T,C]
  u16* Vtb  = (u16*)alloc((size_t)BB * HH * CC * TT * 2);     // [B,H,C,T]  (V transposed)
  u16* S    = (u16*)alloc((size_t)BB * HH * TT * TT * 2);     // scores/probs [B,H,T,T]
  u16* attb = (u16*)alloc((size_t)BB * TT * HH * CC * 2);     // cat layout [B,T,H*C]
  u16* h2b  = (u16*)alloc((size_t)BB * TT * CC * 2);          // ln2(x2) bf16
  u16* ff1b = (u16*)alloc((size_t)BB * TT * 4 * CC * 2);      // relu(h2 W1^T + b1)

  dim3 blk(256);
  auto conv = [&](const float* src, u16* dst, int n) {
    k_f32_to_bf16<<<dim3((n + 255) / 256), blk, 0, stream>>>(src, dst, n);
  };
  conv(Wq, wqb, HH * CC * CC);
  conv(Wk, wkb, HH * CC * CC);
  conv(Wv, wvb, HH * CC * CC);
  conv(Wo, wob, CC * HH * CC);
  conv(W1, w1b, 4 * CC * CC);
  conv(W2, w2b, CC * 4 * CC);

  auto gemm = [&](const u16* A, int lda, int divA, ll sA1, ll sA2,
                  const u16* Wp, int ldw, int divW, ll sW1, ll sW2,
                  void* O, int ldo, int divO, ll sO1, ll sO2,
                  const float* R, const float* bia,
                  int M, int N, int K, float scale, int flags, int Z) {
    dim3 grid(N / 128, M / 64, Z);
    k_gemm_bf16<<<grid, blk, 0, stream>>>(A, lda, divA, sA1, sA2,
                                          Wp, ldw, divW, sW1, sW2,
                                          O, ldo, divO, sO1, sO2,
                                          R, bia, M, N, K, scale, flags);
  };

  const ll TC = (ll)TT * CC, CCs = (ll)CC * CC, TTs = (ll)TT * TT;

  // h = ln1(x)
  k_layernorm_bf16<<<dim3(BB * TT), blk, 0, stream>>>(x, g1, be1, hb);

  // Q/K/V: per (b,h) GEMM [T,C]x[C,C]^T ; z = b*H + h ; A depends on b, W on h.
  gemm(hb, CC, HH, TC, 0,  wqb, CC, HH, 0, CCs,  Qb,  CC, 1, TC, 0,
       nullptr, nullptr, TT, CC, CC, 1.0f, F_OBF16, BB * HH);
  gemm(hb, CC, HH, TC, 0,  wkb, CC, HH, 0, CCs,  Kb,  CC, 1, TC, 0,
       nullptr, nullptr, TT, CC, CC, 1.0f, F_OBF16, BB * HH);
  gemm(hb, CC, HH, TC, 0,  wvb, CC, HH, 0, CCs,  Vtb, TT, 1, (ll)CC * TT, 0,
       nullptr, nullptr, TT, CC, CC, 1.0f, F_OBF16 | F_TRANS, BB * HH);

  // scores = Q K^T / sqrt(C)  (causal: skip upper-triangle blocks)
  gemm(Qb, CC, 1, TC, 0,  Kb, CC, 1, TC, 0,  S, TT, 1, TTs, 0,
       nullptr, nullptr, TT, TT, CC, 0.044194173824159216f, F_OBF16 | F_CAUSAL, BB * HH);

  // probs = causal softmax(scores), zeros above diagonal
  k_softmax_causal<<<dim3(BB * HH * TT), blk, 0, stream>>>(S);

  // att = probs @ V  (W = V^T is [N=C, K=T]); store into cat layout [B,T,H*C]
  gemm(S, TT, 1, TTs, 0,  Vtb, TT, 1, (ll)CC * TT, 0,
       attb, HH * CC, HH, (ll)TT * HH * CC, CC,
       nullptr, nullptr, TT, CC, TT, 1.0f, F_OBF16, BB * HH);

  // x2 = x + cat @ Wo^T + bo  -> write fp32 into d_out
  gemm(attb, HH * CC, 1, 0, 0,  wob, HH * CC, 1, 0, 0,  out, CC, 1, 0, 0,
       x, bo, BB * TT, CC, HH * CC, 1.0f, 0, 1);

  // h2 = ln2(x2)
  k_layernorm_bf16<<<dim3(BB * TT), blk, 0, stream>>>(out, g2, be2, h2b);

  // ff1 = relu(h2 @ W1^T + b1)  bf16
  gemm(h2b, CC, 1, 0, 0,  w1b, CC, 1, 0, 0,  ff1b, 4 * CC, 1, 0, 0,
       nullptr, b1, BB * TT, 4 * CC, CC, 1.0f, F_OBF16 | F_RELU, 1);

  // out = x2 + ff1 @ W2^T + b2   (residual read of d_out, fp32 store)
  gemm(ff1b, 4 * CC, 1, 0, 0,  w2b, 4 * CC, 1, 0, 0,  out, CC, 1, 0, 0,
       out, b2, BB * TT, CC, 4 * CC, 1.0f, 0, 1);
}
